// MambaModel_73650099192141
// MI455X (gfx1250) — matrix operand order repaired and checked by reading, not compiled, not run on hardware
//
#include <hip/hip_runtime.h>
#include <hip/hip_bf16.h>
#include <math.h>

// ---- model dims (match reference) ----
#define BB    2
#define SEQL  1024
#define BDIM  768
#define NST   16
#define FF    3072
#define VV    32000
#define NLAY  4
#define NSUB  3
#define MROWS (BB * SEQL)   // 2048

// ---- WMMA types (CDNA5 wave32, 16x16x32 bf16 -> f32) ----
typedef __bf16 bf16x16 __attribute__((ext_vector_type(16)));
typedef float  f32x8   __attribute__((ext_vector_type(8)));
typedef int    v4i     __attribute__((vector_size(16)));   // matches builtin param type

struct U8x4 { unsigned int u[8]; };

__device__ __forceinline__ unsigned short f32_to_bf16_rne(float f) {
    unsigned int u = __builtin_bit_cast(unsigned int, f);
    u += 0x7FFFu + ((u >> 16) & 1u);          // round-to-nearest-even
    return (unsigned short)(u >> 16);
}

// ---- CDNA5 async global->LDS copy (ASYNCcnt-tracked), with safe fallback ----
#if defined(__has_builtin)
#if __has_builtin(__builtin_amdgcn_global_load_async_to_lds_b128)
#define HAVE_ASYNC_LDS 1
#endif
#endif

#ifdef HAVE_ASYNC_LDS
#define ASYNC_COPY16(gp, lp)                                                   \
    __builtin_amdgcn_global_load_async_to_lds_b128(                            \
        (__attribute__((address_space(1))) v4i*)(gp),                          \
        (__attribute__((address_space(3))) v4i*)(lp), 0, 0)
#if __has_builtin(__builtin_amdgcn_s_wait_asynccnt)
#define WAIT_ASYNC(n) __builtin_amdgcn_s_wait_asynccnt(n)
#else
#define WAIT_ASYNC(n) asm volatile("s_wait_asynccnt %0" ::"i"(n) : "memory")
#endif
#else
#define ASYNC_COPY16(gp, lp) (*(uint4*)(lp) = *(const uint4*)(gp))
#define WAIT_ASYNC(n) ((void)0)
#endif

// =====================================================================
// Tiled bf16 WMMA GEMM, double-buffered async-LDS pipeline.
//   C[M,N] = epilogue(A[M,K] * Bt[N,K]^T + bias[N])
// A: bf16 [M,K] row-major.  Bt: bf16 [N,K] (pre-transposed weights).
// EPI: 0=none 1=gelu(exact) 2=softplus 3=residual add
// OUTBF: 1 -> write bf16 (feeds next GEMM), 0 -> write f32
// Block: 256 threads (8 waves), tile 128x128, BK=32.
// Wave (wm 0..3, wn 0..1) owns 32x64 patch = 2x4 WMMA tiles.
// =====================================================================
template <int EPI, int OUTBF>
__global__ __launch_bounds__(256)
void gemm_wmma_bf16(const unsigned short* __restrict__ A,
                    const unsigned short* __restrict__ Bt,
                    const float* __restrict__ bias, const float* __restrict__ resid,
                    float* __restrict__ C, unsigned short* __restrict__ Cb,
                    int M, int N, int K)
{
    // pitch 40 shorts = 80B: rows 16B-aligned (async B128 stores), and the
    // stride-20-dword rows map 16 consecutive rows onto 16 distinct banks.
    __shared__ unsigned short As[2][128][40];
    __shared__ unsigned short Bs[2][128][40];

    const int t    = threadIdx.x;
    const int lane = t & 31;
    const int wave = t >> 5;
    const int wm   = wave >> 1;              // 0..3
    const int wn   = wave & 1;               // 0..1
    const int m0   = blockIdx.y * 128;
    const int n0   = blockIdx.x * 128;

    f32x8 acc[2][4] = {};

    // one 128x32 bf16 tile = 512 x 16B chunks; 256 threads -> 2 chunks each
    auto load_tiles = [&](int buf, int k0) {
        #pragma unroll
        for (int i = 0; i < 2; ++i) {
            int idx = t + i * 256;           // 0..511
            int row = idx >> 2;              // 4 x 16B per row (32 bf16)
            int q   = (idx & 3) << 3;        // bf16 offset within row
            const unsigned short* gA = &A [(size_t)(m0 + row) * K + k0 + q];
            const unsigned short* gB = &Bt[(size_t)(n0 + row) * K + k0 + q];
            ASYNC_COPY16(gA, &As[buf][row][q]);
            ASYNC_COPY16(gB, &Bs[buf][row][q]);
        }
    };

    const int nk = K >> 5;
    load_tiles(0, 0);

    for (int ks = 0; ks < nk; ++ks) {
        const int cur = ks & 1;
        if (ks + 1 < nk) {
            load_tiles(cur ^ 1, (ks + 1) << 5);
            WAIT_ASYNC(4);                   // stage-k done; next 4 in flight
        } else {
            WAIT_ASYNC(0);
        }
        __syncthreads();

        // ---- fragments per ISA 7.12.2 layouts ----
        bf16x16 afr[2], bfr[4];
        #pragma unroll
        for (int mt = 0; mt < 2; ++mt) {
            int row = wm * 32 + mt * 16 + (lane & 15);
            U8x4 tmp;
            #pragma unroll
            for (int p = 0; p < 8; ++p) {
                // A 16x32 bf16: K = 16*(p>=4) + 8*(lane>=16) + 2*(p%4)
                int k = ((p & 4) << 2) + ((lane & 16) >> 1) + ((p & 3) << 1);
                tmp.u[p] = *reinterpret_cast<const unsigned int*>(&As[cur][row][k]);
            }
            afr[mt] = __builtin_bit_cast(bf16x16, tmp);
        }
        #pragma unroll
        for (int nt = 0; nt < 4; ++nt) {
            int col = wn * 64 + nt * 16 + (lane & 15);
            U8x4 tmp;
            #pragma unroll
            for (int p = 0; p < 8; ++p) {
                // B 32x16 bf16: K = 16*(p>=4) + 8*(lane>=16) + 2*(p%4)
                int k = ((p & 4) << 2) + ((lane & 16) >> 1) + ((p & 3) << 1);
                tmp.u[p] = *reinterpret_cast<const unsigned int*>(&Bs[cur][col][k]);
            }
            bfr[nt] = __builtin_bit_cast(bf16x16, tmp);
        }

        #pragma unroll
        for (int mt = 0; mt < 2; ++mt)
            #pragma unroll
            for (int nt = 0; nt < 4; ++nt)
                acc[mt][nt] = __builtin_amdgcn_wmma_f32_16x16x32_bf16(
                    false, afr[mt], false, bfr[nt],
                    (short)0, acc[mt][nt], false, false);

        __syncthreads();   // protect buf (cur^1 is being filled; cur re-read done)
    }

    // ---- epilogue (C/D layout: VGPR r -> M=r+8*(lane>=16), N=lane%16) ----
    #pragma unroll
    for (int mt = 0; mt < 2; ++mt) {
        #pragma unroll
        for (int nt = 0; nt < 4; ++nt) {
            #pragma unroll
            for (int r = 0; r < 8; ++r) {
                int grow = m0 + wm * 32 + mt * 16 + ((lane >> 4) << 3) + r;
                int gcol = n0 + wn * 64 + nt * 16 + (lane & 15);
                float v = acc[mt][nt][r] + bias[gcol];
                if (EPI == 1) {                       // exact GELU
                    v = 0.5f * v * (1.0f + erff(v * 0.70710678f));
                } else if (EPI == 2) {                // softplus
                    v = (v > 20.0f) ? v : log1pf(__expf(v));
                } else if (EPI == 3) {                // residual add
                    v += resid[(size_t)grow * N + gcol];
                }
                if (OUTBF) Cb[(size_t)grow * N + gcol] = f32_to_bf16_rne(v);
                else       C [(size_t)grow * N + gcol] = v;
            }
        }
    }
}

// =====================================================================
// Weight prep: f32 [K,N] -> bf16 [N,K] (transpose + cast), 32x32 LDS tiles
// =====================================================================
__global__ __launch_bounds__(256)
void transpose_cast_bf16(const float* __restrict__ in, unsigned short* __restrict__ outp,
                         int K, int N)
{
    __shared__ float tile[32][33];
    int kb = blockIdx.y * 32, nb = blockIdx.x * 32;
    int tx = threadIdx.x & 31, ty = threadIdx.x >> 5;   // 8 rows per pass
    for (int r = ty; r < 32; r += 8)
        tile[r][tx] = in[(size_t)(kb + r) * N + (nb + tx)];
    __syncthreads();
    for (int r = ty; r < 32; r += 8)
        outp[(size_t)(nb + r) * K + (kb + tx)] = f32_to_bf16_rne(tile[tx][r]);
}

// elementwise f32 -> bf16
__global__ void cast_bf16_kernel(const float* __restrict__ in,
                                 unsigned short* __restrict__ outp, int n)
{
    int i = blockIdx.x * blockDim.x + threadIdx.x;
    if (i < n) outp[i] = f32_to_bf16_rne(in[i]);
}

// =====================================================================
// Skinny GEMM: out[M,16] = A[M,K] * W[K,16]   (W_B / W_C projections)
// =====================================================================
__global__ __launch_bounds__(256)
void gemm_n16(const float* __restrict__ A, const float* __restrict__ W,
              float* __restrict__ out, int M, int K)
{
    int n   = threadIdx.x & 15;
    int row = blockIdx.x * 16 + (threadIdx.x >> 4);
    if (row >= M) return;
    float s = 0.f;
    for (int k = 0; k < K; ++k)
        s += A[(size_t)row * K + k] * W[k * NST + n];
    out[(size_t)row * NST + n] = s;
}

// =====================================================================
// Causal depthwise conv (k=3) + SiLU; writes f32 (for scan) + bf16 (GEMM A)
// =====================================================================
__global__ void conv_silu_kernel(const float* __restrict__ h,
                                 const float* __restrict__ cw,
                                 float* __restrict__ u,
                                 unsigned short* __restrict__ ubf)
{
    int i = blockIdx.x * blockDim.x + threadIdx.x;
    if (i >= BB * SEQL * BDIM) return;
    int d  = i % BDIM;
    int tl = (i / BDIM) % SEQL;
    float w0 = cw[d * 3 + 0], w1 = cw[d * 3 + 1], w2 = cw[d * 3 + 2];
    float x2 = h[i];
    float x1 = (tl >= 1) ? h[i - BDIM]     : 0.f;
    float x0 = (tl >= 2) ? h[i - 2 * BDIM] : 0.f;
    float v = x0 * w0 + x1 * w1 + x2 * w2;
    v = v * (1.0f / (1.0f + __expf(-v)));          // SiLU
    u[i]   = v;
    ubf[i] = f32_to_bf16_rne(v);
}

// =====================================================================
// Selective scan: one thread per (b,d,n); 16-lane shuffle reduction for y.
// h[b,t,d] += y + D_skip[d]*u   (sublayer residual folded in)
// 24576 threads = 768 wave32 waves of latency-hiding parallelism.
// =====================================================================
__global__ __launch_bounds__(256)
void ssm_scan(const float* __restrict__ dt, const float* __restrict__ u,
              const float* __restrict__ Bm, const float* __restrict__ Cm,
              const float* __restrict__ A_log, const float* __restrict__ Dsk,
              float* __restrict__ h)
{
    int gi = blockIdx.x * blockDim.x + threadIdx.x;   // over BB*BDIM*NST
    if (gi >= BB * BDIM * NST) return;
    int n  = gi & (NST - 1);
    int bd = gi >> 4;
    int b = bd / BDIM, d = bd % BDIM;

    float a   = -__expf(A_log[d * NST + n]);
    float dsk = Dsk[d];
    float s   = 0.f;

    for (int tstep = 0; tstep < SEQL; ++tstep) {
        size_t ro = (size_t)b * SEQL + tstep;
        float dtv = dt[ro * BDIM + d];     // broadcast across the 16 n-lanes
        float uv  = u [ro * BDIM + d];
        s = __expf(dtv * a) * s + (dtv * uv) * Bm[ro * NST + n];
        float y = s * Cm[ro * NST + n];
        y += __shfl_xor(y, 1, 32);         // reduce over n within 16-lane group
        y += __shfl_xor(y, 2, 32);
        y += __shfl_xor(y, 4, 32);
        y += __shfl_xor(y, 8, 32);
        if (n == 0) h[ro * BDIM + d] += y + dsk * uv;
    }
}

// =====================================================================
// Embedding gather
// =====================================================================
__global__ void embed_kernel(const int* __restrict__ x,
                             const float* __restrict__ emb,
                             float* __restrict__ h)
{
    int i = blockIdx.x * blockDim.x + threadIdx.x;
    if (i >= BB * SEQL * BDIM) return;
    int row = i / BDIM, d = i % BDIM;
    h[i] = emb[(size_t)x[row] * BDIM + d];
}

// =====================================================================
// LayerNorm over D=768; one block per row; bf16 out (feeds final GEMM)
// =====================================================================
__global__ __launch_bounds__(256)
void layernorm_kernel(const float* __restrict__ in, const float* __restrict__ g,
                      const float* __restrict__ bb, unsigned short* __restrict__ outbf)
{
    __shared__ float red[256];
    int row = blockIdx.x, tid = threadIdx.x;
    const float* x = in + (size_t)row * BDIM;

    float s = 0.f;
    for (int d = tid; d < BDIM; d += 256) s += x[d];
    red[tid] = s; __syncthreads();
    for (int o = 128; o > 0; o >>= 1) {
        if (tid < o) red[tid] += red[tid + o];
        __syncthreads();
    }
    float mean = red[0] / BDIM;
    __syncthreads();

    float v = 0.f;
    for (int d = tid; d < BDIM; d += 256) { float t2 = x[d] - mean; v += t2 * t2; }
    red[tid] = v; __syncthreads();
    for (int o = 128; o > 0; o >>= 1) {
        if (tid < o) red[tid] += red[tid + o];
        __syncthreads();
    }
    float rstd = rsqrtf(red[0] / BDIM + 1e-5f);

    for (int d = tid; d < BDIM; d += 256)
        outbf[(size_t)row * BDIM + d] =
            f32_to_bf16_rne((x[d] - mean) * rstd * g[d] + bb[d]);
}

// =====================================================================
extern "C" void kernel_launch(void* const* d_in, const int* in_sizes, int n_in,
                              void* d_out, int out_size, void* d_ws, size_t ws_size,
                              hipStream_t stream)
{
    const int*   x       = (const int*)  d_in[0];
    const float* emb     = (const float*)d_in[1];
    const float* conv_w  = (const float*)d_in[2];
    const float* A_log   = (const float*)d_in[3];
    const float* W_B     = (const float*)d_in[4];
    const float* W_C     = (const float*)d_in[5];
    const float* W_dt    = (const float*)d_in[6];
    const float* dt_bias = (const float*)d_in[7];
    const float* D_skip  = (const float*)d_in[8];
    const float* ffn_W1  = (const float*)d_in[9];
    const float* ffn_b1  = (const float*)d_in[10];
    const float* ffn_W2  = (const float*)d_in[11];
    const float* ffn_b2  = (const float*)d_in[12];
    const float* norm_g  = (const float*)d_in[13];
    const float* norm_b  = (const float*)d_in[14];
    const float* out_W   = (const float*)d_in[15];
    const float* out_b   = (const float*)d_in[16];
    float* out = (float*)d_out;

    const size_t MD = (size_t)MROWS * BDIM;

    // byte-based workspace allocator (256B aligned blocks)
    char* wp = (char*)d_ws;
    auto alloc = [&](size_t bytes) -> void* {
        void* r = (void*)wp;
        wp += (bytes + 255) & ~(size_t)255;
        return r;
    };
    float*          h      = (float*)         alloc(MD * 4);
    float*          u      = (float*)         alloc(MD * 4);
    float*          dtv    = (float*)         alloc(MD * 4);
    float*          Bm     = (float*)         alloc((size_t)MROWS * NST * 4);
    float*          Cm     = (float*)         alloc((size_t)MROWS * NST * 4);
    unsigned short* u_bf   = (unsigned short*)alloc(MD * 2);
    unsigned short* h_bf   = (unsigned short*)alloc(MD * 2);
    unsigned short* f_bf   = (unsigned short*)alloc((size_t)MROWS * FF * 2);
    unsigned short* nrm_bf = (unsigned short*)alloc(MD * 2);
    unsigned short* Wt_dt  = (unsigned short*)alloc((size_t)NLAY * NSUB * BDIM * BDIM * 2);
    unsigned short* Wt_1   = (unsigned short*)alloc((size_t)NLAY * BDIM * FF * 2);
    unsigned short* Wt_2   = (unsigned short*)alloc((size_t)NLAY * FF * BDIM * 2);
    unsigned short* Wt_out = (unsigned short*)alloc((size_t)BDIM * VV * 2);

    const int elems = MROWS * BDIM;
    const dim3 blk(256);

    // ---- one-time weight prep: transpose+cast to bf16 [N,K] ----
    for (int ls = 0; ls < NLAY * NSUB; ++ls)
        transpose_cast_bf16<<<dim3(BDIM / 32, BDIM / 32), blk, 0, stream>>>(
            W_dt + (size_t)ls * BDIM * BDIM, Wt_dt + (size_t)ls * BDIM * BDIM, BDIM, BDIM);
    for (int l = 0; l < NLAY; ++l) {
        transpose_cast_bf16<<<dim3(FF / 32, BDIM / 32), blk, 0, stream>>>(
            ffn_W1 + (size_t)l * BDIM * FF, Wt_1 + (size_t)l * BDIM * FF, BDIM, FF);
        transpose_cast_bf16<<<dim3(BDIM / 32, FF / 32), blk, 0, stream>>>(
            ffn_W2 + (size_t)l * FF * BDIM, Wt_2 + (size_t)l * FF * BDIM, FF, BDIM);
    }
    transpose_cast_bf16<<<dim3(VV / 32, BDIM / 32), blk, 0, stream>>>(
        out_W, Wt_out, BDIM, VV);

    embed_kernel<<<(elems + 255) / 256, blk, 0, stream>>>(x, emb, h);

    for (int l = 0; l < NLAY; ++l) {
        for (int s = 0; s < NSUB; ++s) {
            size_t ls = (size_t)(l * NSUB + s);
            conv_silu_kernel<<<(elems + 255) / 256, blk, 0, stream>>>(
                h, conv_w + ls * BDIM * 3, u, u_bf);

            // dt = softplus(u @ W_dt + dt_bias)
            gemm_wmma_bf16<2, 0><<<dim3(BDIM / 128, MROWS / 128), blk, 0, stream>>>(
                u_bf, Wt_dt + ls * BDIM * BDIM, dt_bias + ls * BDIM, nullptr,
                dtv, nullptr, MROWS, BDIM, BDIM);

            gemm_n16<<<MROWS / 16, blk, 0, stream>>>(u, W_B + ls * BDIM * NST, Bm, MROWS, BDIM);
            gemm_n16<<<MROWS / 16, blk, 0, stream>>>(u, W_C + ls * BDIM * NST, Cm, MROWS, BDIM);

            ssm_scan<<<(BB * BDIM * NST + 255) / 256, blk, 0, stream>>>(
                dtv, u, Bm, Cm, A_log + ls * BDIM * NST, D_skip + ls * BDIM, h);
        }
        // f_bf = bf16(gelu(h @ W1 + b1))
        cast_bf16_kernel<<<(elems + 255) / 256, blk, 0, stream>>>(h, h_bf, elems);
        gemm_wmma_bf16<1, 1><<<dim3(FF / 128, MROWS / 128), blk, 0, stream>>>(
            h_bf, Wt_1 + (size_t)l * BDIM * FF, ffn_b1 + (size_t)l * FF, nullptr,
            nullptr, f_bf, MROWS, FF, BDIM);
        // h = h + f @ W2 + b2
        gemm_wmma_bf16<3, 0><<<dim3(BDIM / 128, MROWS / 128), blk, 0, stream>>>(
            f_bf, Wt_2 + (size_t)l * FF * BDIM, ffn_b2 + (size_t)l * BDIM, h,
            h, nullptr, MROWS, BDIM, FF);
    }

    layernorm_kernel<<<MROWS, blk, 0, stream>>>(h, norm_g, norm_b, nrm_bf);

    // logits = nrm @ out_W + out_b  (out_W bf16: 49MB -> fully L2-resident)
    gemm_wmma_bf16<0, 0><<<dim3(VV / 128, MROWS / 128), blk, 0, stream>>>(
        nrm_bf, Wt_out, out_b, nullptr, out, nullptr, MROWS, VV, BDIM);
}
